// U_Transformer_85693187490476
// MI455X (gfx1250) — matrix operand order repaired and checked by reading, not compiled, not run on hardware
//
#include <hip/hip_runtime.h>
#include <hip/hip_bf16.h>
#include <math.h>

typedef __bf16 bf16_t;
typedef __attribute__((ext_vector_type(8)))  __bf16 v8bf;
typedef __attribute__((ext_vector_type(16))) __bf16 v16bf;
typedef __attribute__((ext_vector_type(8)))  float  v8f;

__device__ __forceinline__ bf16_t f2bf(float f) {
    unsigned u = __builtin_bit_cast(unsigned, f);
    unsigned r = u + 0x7FFFu + ((u >> 16) & 1u);      // round-to-nearest-even
    unsigned short h = (unsigned short)(r >> 16);
    return __builtin_bit_cast(bf16_t, h);
}

// ---------------------------------------------------------------------------
// fp32 -> bf16 elementwise convert
// ---------------------------------------------------------------------------
__global__ void __launch_bounds__(256) f32_to_bf16_k(const float* __restrict__ x,
                                                     bf16_t* __restrict__ y, int n) {
    int i = blockIdx.x * 256 + threadIdx.x;
    if (i < n) y[i] = f2bf(x[i]);
}

// ---------------------------------------------------------------------------
// NT-GEMM mainloop: D(64x64 block) += A[m][k] * Bt[n][k], bf16 in, f32 acc.
// 64 threads = 2 waves; each wave owns a 32x64 tile (2x4 subtiles of 16x16).
// Software-pipelined with 8 *scalar* uint4 registers (no arrays -> guaranteed
// VGPR promotion): next K-tile loads overlap the current tile's 8 WMMAs.
// ---------------------------------------------------------------------------
__device__ __forceinline__ void gemm_mainloop(const bf16_t* __restrict__ A, int lda,
                                              const bf16_t* __restrict__ Bt, int ldb,
                                              int Kdim, int m0, int n0,
                                              bf16_t* As, bf16_t* Bs, v8f (&acc)[2][4]) {
    const int tid   = threadIdx.x;       // 0..63
    const int wm    = (tid >> 5) * 32;   // wave 0: rows 0-31, wave 1: rows 32-63
    const int lane  = tid & 31;
    const int lhalf = lane >> 4;         // 0: lanes 0-15, 1: lanes 16-31
    const int lrow  = lane & 15;

    const bf16_t* gA = A + (size_t)(m0 + tid) * lda;
    const bf16_t* gB = Bt + (size_t)(n0 + tid) * ldb;
    uint4* sa = reinterpret_cast<uint4*>(As + tid * 32);
    uint4* sb = reinterpret_cast<uint4*>(Bs + tid * 32);

    // prologue: load K-tile 0 into 8 scalar register quads
    uint4 a0, a1, a2, a3, b0, b1, b2, b3;
    {
        const uint4* ga = reinterpret_cast<const uint4*>(gA);
        const uint4* gb = reinterpret_cast<const uint4*>(gB);
        a0 = ga[0]; a1 = ga[1]; a2 = ga[2]; a3 = ga[3];
        b0 = gb[0]; b1 = gb[1]; b2 = gb[2]; b3 = gb[3];
    }

    for (int k0 = 0; k0 < Kdim; k0 += 32) {
        __syncthreads();                 // all frag reads of previous tile done
        sa[0] = a0; sa[1] = a1; sa[2] = a2; sa[3] = a3;
        sb[0] = b0; sb[1] = b1; sb[2] = b2; sb[3] = b3;
        if (k0 + 32 < Kdim) {            // issue next tile's loads now
            const uint4* ga = reinterpret_cast<const uint4*>(gA + k0 + 32);
            const uint4* gb = reinterpret_cast<const uint4*>(gB + k0 + 32);
            a0 = ga[0]; a1 = ga[1]; a2 = ga[2]; a3 = ga[3];
            b0 = gb[0]; b1 = gb[1]; b2 = gb[2]; b3 = gb[3];
            if (k0 + 64 < Kdim) {        // and prefetch the tile after next
                __builtin_prefetch((const void*)(gA + k0 + 64), 0, 3);
                __builtin_prefetch((const void*)(gB + k0 + 64), 0, 3);
            }
        }
        __syncthreads();

        v16bf afrag[2], bfrag[4];
#pragma unroll
        for (int i = 0; i < 2; ++i) {
            // A 16x32 fragment: lane half 0 holds K {0-7,16-23}, half 1 {8-15,24-31}
            const v8bf* ap = reinterpret_cast<const v8bf*>(As + (wm + i * 16 + lrow) * 32);
            v8bf alo = ap[lhalf];
            v8bf ahi = ap[2 + lhalf];
            afrag[i] = __builtin_shufflevector(alo, ahi,
                        0,1,2,3,4,5,6,7,8,9,10,11,12,13,14,15);
        }
#pragma unroll
        for (int j = 0; j < 4; ++j) {
            // B 32x16 fragment: lane = column, lane half 0 holds K {0-7,16-23}, half 1 {8-15,24-31}
            const v8bf* bp = reinterpret_cast<const v8bf*>(Bs + (j * 16 + lrow) * 32);
            v8bf blo = bp[lhalf];
            v8bf bhi = bp[2 + lhalf];
            bfrag[j] = __builtin_shufflevector(blo, bhi,
                        0,1,2,3,4,5,6,7,8,9,10,11,12,13,14,15);
        }
#pragma unroll
        for (int i = 0; i < 2; ++i)
#pragma unroll
            for (int j = 0; j < 4; ++j)
                acc[i][j] = __builtin_amdgcn_wmma_f32_16x16x32_bf16(
                    false, afrag[i], false, bfrag[j], (short)0, acc[i][j], false, false);
    }
}

__device__ __forceinline__ void store_tile_bf16(bf16_t* D, int ldd, int m0, int n0,
                                                const v8f (&acc)[2][4], float scale) {
    const int tid = threadIdx.x;
    const int wm = (tid >> 5) * 32;
    const int lane = tid & 31, lhalf = lane >> 4, lrow = lane & 15;
#pragma unroll
    for (int i = 0; i < 2; ++i)
#pragma unroll
        for (int j = 0; j < 4; ++j) {
            int col = n0 + j * 16 + lrow;
            int rowb = m0 + wm + i * 16 + lhalf * 8;
#pragma unroll
            for (int r = 0; r < 8; ++r)
                D[(size_t)(rowb + r) * ldd + col] = f2bf(acc[i][j][r] * scale);
        }
}

__device__ __forceinline__ void store_tile_f32(float* D, int ldd, int m0, int n0,
                                               const v8f (&acc)[2][4], float scale) {
    const int tid = threadIdx.x;
    const int wm = (tid >> 5) * 32;
    const int lane = tid & 31, lhalf = lane >> 4, lrow = lane & 15;
#pragma unroll
    for (int i = 0; i < 2; ++i)
#pragma unroll
        for (int j = 0; j < 4; ++j) {
            int col = n0 + j * 16 + lrow;
            int rowb = m0 + wm + i * 16 + lhalf * 8;
#pragma unroll
            for (int r = 0; r < 8; ++r)
                D[(size_t)(rowb + r) * ldd + col] = acc[i][j][r] * scale;
        }
}

// ---------------------------------------------------------------------------
// Generic batched NT GEMMs. z = b*Hdim + h; per-tensor (batch, head) strides.
// ---------------------------------------------------------------------------
__global__ void __launch_bounds__(64)
gemm_nt_bf16_kernel(const bf16_t* __restrict__ A, int lda, long long sAb, long long sAh,
                    const bf16_t* __restrict__ Bt, int ldb, long long sBb, long long sBh,
                    bf16_t* __restrict__ D, int ldd, long long sDb, long long sDh,
                    int Kdim, int Hdim, float scale) {
    __shared__ __align__(16) bf16_t As[64 * 32];
    __shared__ __align__(16) bf16_t Bs[64 * 32];
    int z = blockIdx.z, b = z / Hdim, h = z % Hdim;
    const bf16_t* Ab = A + (long long)b * sAb + (long long)h * sAh;
    const bf16_t* Bb = Bt + (long long)b * sBb + (long long)h * sBh;
    bf16_t* Db = D + (long long)b * sDb + (long long)h * sDh;
    v8f acc[2][4];
    v8f zf = {};
#pragma unroll
    for (int i = 0; i < 2; ++i)
#pragma unroll
        for (int j = 0; j < 4; ++j) acc[i][j] = zf;
    int m0 = blockIdx.x * 64, n0 = blockIdx.y * 64;
    gemm_mainloop(Ab, lda, Bb, ldb, Kdim, m0, n0, As, Bs, acc);
    store_tile_bf16(Db, ldd, m0, n0, acc, scale);
}

__global__ void __launch_bounds__(64)
gemm_nt_f32_kernel(const bf16_t* __restrict__ A, int lda, long long sAb, long long sAh,
                   const bf16_t* __restrict__ Bt, int ldb, long long sBb, long long sBh,
                   float* __restrict__ D, int ldd, long long sDb, long long sDh,
                   int Kdim, int Hdim, float scale) {
    __shared__ __align__(16) bf16_t As[64 * 32];
    __shared__ __align__(16) bf16_t Bs[64 * 32];
    int z = blockIdx.z, b = z / Hdim, h = z % Hdim;
    const bf16_t* Ab = A + (long long)b * sAb + (long long)h * sAh;
    const bf16_t* Bb = Bt + (long long)b * sBb + (long long)h * sBh;
    float* Db = D + (long long)b * sDb + (long long)h * sDh;
    v8f acc[2][4];
    v8f zf = {};
#pragma unroll
    for (int i = 0; i < 2; ++i)
#pragma unroll
        for (int j = 0; j < 4; ++j) acc[i][j] = zf;
    int m0 = blockIdx.x * 64, n0 = blockIdx.y * 64;
    gemm_mainloop(Ab, lda, Bb, ldb, Kdim, m0, n0, As, Bs, acc);
    store_tile_f32(Db, ldd, m0, n0, acc, scale);
}

// ctx[b][n][c] = 0.25 * sum_h sum_kv probs[b,h][c][kv] * V[b,h][n][kv]
__global__ void __launch_bounds__(64)
gemm_ctx_kernel(const bf16_t* __restrict__ V, const bf16_t* __restrict__ P,
                bf16_t* __restrict__ D, int C) {
    __shared__ __align__(16) bf16_t As[64 * 32];
    __shared__ __align__(16) bf16_t Bs[64 * 32];
    int b = blockIdx.z;
    int m0 = blockIdx.x * 64;   // token dim (1024)
    int n0 = blockIdx.y * 64;   // channel dim (C)
    v8f acc[2][4];
    v8f zf = {};
#pragma unroll
    for (int i = 0; i < 2; ++i)
#pragma unroll
        for (int j = 0; j < 4; ++j) acc[i][j] = zf;
    for (int h = 0; h < 4; ++h) {
        const bf16_t* Ah = V + (size_t)(b * 4 + h) * 1024 * 960;
        const bf16_t* Bh = P + (size_t)(b * 4 + h) * C * 960;
        gemm_mainloop(Ah, 960, Bh, 960, 960, m0, n0, As, Bs, acc);
    }
    store_tile_bf16(D + (size_t)b * 1024 * C, C, m0, n0, acc, 0.25f);
}

// ---------------------------------------------------------------------------
// Instance-norm stats: one block per (b,h); n = C*960 elements
// ---------------------------------------------------------------------------
__global__ void __launch_bounds__(256) in_stats_kernel(const float* __restrict__ S,
                                                       int n, float* __restrict__ stats) {
    int z = blockIdx.x, tid = threadIdx.x;
    const float* p = S + (size_t)z * n;
    float s = 0.f, s2 = 0.f;
    for (int i = tid; i < n; i += 256) { float v = p[i]; s += v; s2 += v * v; }
    __shared__ float rs[256], rq[256];
    rs[tid] = s; rq[tid] = s2;
    __syncthreads();
    for (int off = 128; off > 0; off >>= 1) {
        if (tid < off) { rs[tid] += rs[tid + off]; rq[tid] += rq[tid + off]; }
        __syncthreads();
    }
    if (tid == 0) {
        float mean = rs[0] / (float)n;
        float var  = rq[0] / (float)n - mean * mean;
        stats[2 * z]     = mean;
        stats[2 * z + 1] = rsqrtf(var + 1e-5f);
    }
}

// ---------------------------------------------------------------------------
// Row softmax over KV=960 of (s-mean)*rstd; one block per (b,h,c) row
// ---------------------------------------------------------------------------
__global__ void __launch_bounds__(256) softmax_row_kernel(const float* __restrict__ S,
                                                          const float* __restrict__ stats,
                                                          bf16_t* __restrict__ P, int C) {
    int row = blockIdx.x, tid = threadIdx.x;
    int z = row / C;
    const float* s = S + (size_t)row * 960;
    float mean = stats[2 * z], rstd = stats[2 * z + 1];
    float vals[4];
    float mx = -1e30f;
    int cnt = 0;
    for (int k = tid; k < 960; k += 256) {
        float y = (s[k] - mean) * rstd;
        vals[cnt++] = y;
        mx = fmaxf(mx, y);
    }
    __shared__ float red[256];
    red[tid] = mx;
    __syncthreads();
    for (int off = 128; off > 0; off >>= 1) {
        if (tid < off) red[tid] = fmaxf(red[tid], red[tid + off]);
        __syncthreads();
    }
    mx = red[0];
    __syncthreads();
    float sum = 0.f;
    for (int i = 0; i < cnt; ++i) { vals[i] = __expf(vals[i] - mx); sum += vals[i]; }
    red[tid] = sum;
    __syncthreads();
    for (int off = 128; off > 0; off >>= 1) {
        if (tid < off) red[tid] += red[tid + off];
        __syncthreads();
    }
    float inv = 1.f / red[0];
    cnt = 0;
    for (int k = tid; k < 960; k += 256)
        P[(size_t)row * 960 + k] = f2bf(vals[cnt++] * inv);
}

// ---------------------------------------------------------------------------
extern "C" void kernel_launch(void* const* d_in, const int* in_sizes, int n_in,
                              void* d_out, int out_size, void* d_ws, size_t ws_size,
                              hipStream_t stream) {
    (void)in_sizes; (void)n_in; (void)out_size; (void)ws_size;
    const int B = 8, N = 1024, H = 4, KV = 960;
    const int Cs[4] = {64, 128, 256, 512};

    const float* emb_f[4] = {(const float*)d_in[0], (const float*)d_in[1],
                             (const float*)d_in[2], (const float*)d_in[3]};
    const float* embA_f = (const float*)d_in[4];
    const float* Wq_f[4] = {(const float*)d_in[5], (const float*)d_in[6],
                            (const float*)d_in[7], (const float*)d_in[8]};
    const float* Wo_f[4] = {(const float*)d_in[9], (const float*)d_in[10],
                            (const float*)d_in[11], (const float*)d_in[12]};
    const float* Wk_f = (const float*)d_in[13];
    const float* Wv_f = (const float*)d_in[14];
    float* out = (float*)d_out;

    size_t off = 0;
    auto take = [&](size_t bytes) -> char* {
        char* p = (char*)d_ws + off;
        off += (bytes + 255) & ~(size_t)255;
        return p;
    };
    bf16_t* embA = (bf16_t*)take((size_t)B * N * KV * 2);
    bf16_t* Wkb  = (bf16_t*)take((size_t)H * KV * KV * 2);
    bf16_t* Wvb  = (bf16_t*)take((size_t)H * KV * KV * 2);
    bf16_t* KT   = (bf16_t*)take((size_t)B * H * KV * N * 2);   // [b,h][kv][n]
    bf16_t* Vb   = (bf16_t*)take((size_t)B * H * N * KV * 2);   // [b,h][n][kv]
    bf16_t* embb[4]; bf16_t* Wqb[4]; bf16_t* Wob[4];
    for (int i = 0; i < 4; ++i) embb[i] = (bf16_t*)take((size_t)B * N * Cs[i] * 2);
    for (int i = 0; i < 4; ++i) Wqb[i]  = (bf16_t*)take((size_t)H * Cs[i] * Cs[i] * 2);
    for (int i = 0; i < 4; ++i) Wob[i]  = (bf16_t*)take((size_t)Cs[i] * Cs[i] * 2);
    bf16_t* QT  = (bf16_t*)take((size_t)B * H * 512 * N * 2);   // [b,h][c][n] (reused)
    float*  Sm  = (float*) take((size_t)B * H * 512 * KV * 4);  // scores (reused)
    bf16_t* Pm  = (bf16_t*)take((size_t)B * H * 512 * KV * 2);  // probs (reused)
    bf16_t* CTX = (bf16_t*)take((size_t)B * N * 512 * 2);       // ctx (reused)
    float*  stats = (float*)take((size_t)2 * B * H * 4);

    auto conv = [&](const float* x, bf16_t* y, size_t n) {
        f32_to_bf16_k<<<dim3((unsigned)((n + 255) / 256)), dim3(256), 0, stream>>>(x, y, (int)n);
    };
    conv(embA_f, embA, (size_t)B * N * KV);
    conv(Wk_f, Wkb, (size_t)H * KV * KV);
    conv(Wv_f, Wvb, (size_t)H * KV * KV);
    for (int i = 0; i < 4; ++i) {
        conv(emb_f[i], embb[i], (size_t)B * N * Cs[i]);
        conv(Wq_f[i], Wqb[i], (size_t)H * Cs[i] * Cs[i]);
        conv(Wo_f[i], Wob[i], (size_t)Cs[i] * Cs[i]);
    }

    const float inv_sqrt_kv = 1.0f / sqrtf((float)KV);

    // KT[b,h][kv][n] = sum_k Wk[h][kv][k] * emb_all[b][n][k]   (M=960,N=1024,K=960)
    gemm_nt_bf16_kernel<<<dim3(KV / 64, N / 64, B * H), dim3(64), 0, stream>>>(
        Wkb, KV, 0LL, (long long)KV * KV,
        embA, KV, (long long)N * KV, 0LL,
        KT, N, (long long)H * KV * N, (long long)KV * N,
        KV, H, 1.0f);
    // V[b,h][n][kv] = sum_k emb_all[b][n][k] * Wv[h][kv][k]    (M=1024,N=960,K=960)
    gemm_nt_bf16_kernel<<<dim3(N / 64, KV / 64, B * H), dim3(64), 0, stream>>>(
        embA, KV, (long long)N * KV, 0LL,
        Wvb, KV, 0LL, (long long)KV * KV,
        Vb, KV, (long long)H * N * KV, (long long)N * KV,
        KV, H, 1.0f);

    long long outOff = 0;
    for (int i = 0; i < 4; ++i) {
        const int C = Cs[i];
        // QT[b,h][c][n] = sum_k Wq[h][c][k] * emb[b][n][k]      (M=C,N=1024,K=C)
        gemm_nt_bf16_kernel<<<dim3(C / 64, N / 64, B * H), dim3(64), 0, stream>>>(
            Wqb[i], C, 0LL, (long long)C * C,
            embb[i], C, (long long)N * C, 0LL,
            QT, N, (long long)H * C * N, (long long)C * N,
            C, H, 1.0f);
        // scores[b,h][c][kv] = (1/sqrt(KV)) sum_n QT[c][n]*KT[kv][n]  (M=C,N=960,K=1024)
        gemm_nt_f32_kernel<<<dim3(C / 64, KV / 64, B * H), dim3(64), 0, stream>>>(
            QT, N, (long long)H * C * N, (long long)C * N,
            KT, N, (long long)H * KV * N, (long long)KV * N,
            Sm, KV, (long long)H * C * KV, (long long)C * KV,
            N, H, inv_sqrt_kv);
        // instance-norm stats over (C,KV) per (b,h)
        in_stats_kernel<<<dim3(B * H), dim3(256), 0, stream>>>(Sm, C * KV, stats);
        // softmax over KV per row -> bf16 probs
        softmax_row_kernel<<<dim3(B * H * C), dim3(256), 0, stream>>>(Sm, stats, Pm, C);
        // ctx[b][n][c] = mean_h probs[c][kv]*V[n][kv]           (M=1024,N=C,K=960, 4 heads)
        gemm_ctx_kernel<<<dim3(N / 64, C / 64, B), dim3(64), 0, stream>>>(Vb, Pm, CTX, C);
        // out[b][n][o] = sum_c ctx[n][c]*Wo[o][c]               (M=1024,N=C,K=C)
        gemm_nt_f32_kernel<<<dim3(N / 64, C / 64, B), dim3(64), 0, stream>>>(
            CTX, C, (long long)N * C, 0LL,
            Wob[i], C, 0LL, 0LL,
            out + outOff, C, (long long)N * C, 0LL,
            C, 1, 1.0f);
        outOff += (long long)B * N * C;
    }
}
